// MultiheadAttention_42829413875998
// MI455X (gfx1250) — matrix operand order repaired and checked by reading, not compiled, not run on hardware
//
#include <hip/hip_runtime.h>
#include <hip/hip_bf16.h>

typedef __attribute__((ext_vector_type(16))) __bf16 v16bf;
typedef __attribute__((ext_vector_type(8)))  __bf16 v8bf;
typedef __attribute__((ext_vector_type(8)))  float  v8f;

#define T_LEN 2048
#define BSZ   4
#define EMB   512
#define NH    8
#define HD    64
#define NBH   32      // BSZ*NH
#define MROWS 8192    // T_LEN*BSZ
#define QKSCALE 0.125f
#define NEG_BIG -1.0e9f

static __device__ __forceinline__ __bf16 f2bf(float x) {
  unsigned u = __builtin_bit_cast(unsigned, x);
  unsigned r = (u + 0x7FFFu + ((u >> 16) & 1u)) >> 16;
  unsigned short s = (unsigned short)r;
  return __builtin_bit_cast(__bf16, s);
}

static __device__ __forceinline__ v16bf cat16(v8bf lo, v8bf hi) {
  return __builtin_shufflevector(lo, hi, 0,1,2,3,4,5,6,7,8,9,10,11,12,13,14,15);
}

// A-operand fragment per ISA 16-bit A layout (16xK tile, K-chunk of 32):
// lane l (half=l>>4) holds row m0+(l&15); VGPR pairs cover K = {k0+8*half..+7} and {k0+16+8*half..+7}
static __device__ __forceinline__ v16bf load_a(const __bf16* A, int ld, int m0, int k0, int lane) {
  int half = lane >> 4;
  const __bf16* p = A + (size_t)(m0 + (lane & 15)) * ld + k0 + half * 8;
  v8bf lo = *(const v8bf*)p;
  v8bf hi = *(const v8bf*)(p + 16);
  return cat16(lo, hi);
}

// B-operand fragment: lane l holds output col n0+(l&15), K = {k0+8*half..+7} and {k0+16+8*half..+7}
static __device__ __forceinline__ v16bf load_b(const __bf16* Bm, int ld, int n0, int k0, int lane) {
  int half = lane >> 4;
  const __bf16* p = Bm + (size_t)(n0 + (lane & 15)) * ld + k0 + half * 8;
  v8bf lo = *(const v8bf*)p;
  v8bf hi = *(const v8bf*)(p + 16);
  return cat16(lo, hi);
}

static __device__ __forceinline__ v8f wmma_bf16(v16bf a, v16bf b, v8f c) {
  return __builtin_amdgcn_wmma_f32_16x16x32_bf16(false, a, false, b, (short)0, c, false, false);
}

// ---------------- fp32 -> bf16 convert ----------------
__global__ void mha_cvt_bf16(const float* __restrict__ src, __bf16* __restrict__ dst, int n) {
  int i = (blockIdx.x * blockDim.x + threadIdx.x) * 4;
  if (i + 3 < n) {
    float4 v = *(const float4*)(src + i);
    dst[i + 0] = f2bf(v.x);
    dst[i + 1] = f2bf(v.y);
    dst[i + 2] = f2bf(v.z);
    dst[i + 3] = f2bf(v.w);
  }
}

// ---------------- fused QKV projection ----------------
// qkv[m, f] = sum_e hb[m,e] * Wqkv[f,e] + bias[f];  m = t*BSZ + b
// Epilogue scatters into head-major bf16 buffers:
//   Q (scaled) -> Qh[bh, t, d], K -> Kh[bh, t, d], V -> Vt[bh, d, t] (transposed!)
__global__ void mha_qkv_gemm(const __bf16* __restrict__ hb, const __bf16* __restrict__ wq,
                             const float* __restrict__ bias,
                             __bf16* __restrict__ Qh, __bf16* __restrict__ Kh,
                             __bf16* __restrict__ Vt) {
  int lane = threadIdx.x & 31;
  int wave = threadIdx.x >> 5;
  int m0 = blockIdx.x * 16;
  int n0 = (blockIdx.y * 4 + wave) * 16;

  v8f c = {};
  for (int k0 = 0; k0 < EMB; k0 += 32) {
    v16bf a = load_a(hb, EMB, m0, k0, lane);
    v16bf b = load_b(wq, EMB, n0, k0, lane);
    c = wmma_bf16(a, b, c);
  }

  int half = lane >> 4, nl = lane & 15;
  int f = n0 + nl;
  float bv = bias[f];
  int which = f >> 9;         // 0=Q,1=K,2=V
  int e = f & 511;
  int hh = e >> 6, d = e & 63;
#pragma unroll
  for (int r = 0; r < 8; ++r) {
    int mrow = m0 + r + 8 * half;
    int t = mrow >> 2, b_ = mrow & 3;
    int bh = b_ * NH + hh;
    float v = c[r] + bv;
    if (which == 0)      Qh[((size_t)bh * T_LEN + t) * HD + d] = f2bf(v * QKSCALE);
    else if (which == 1) Kh[((size_t)bh * T_LEN + t) * HD + d] = f2bf(v);
    else                 Vt[((size_t)bh * HD + d) * T_LEN + t] = f2bf(v);
  }
}

// ---------------- causal attention: softmax stats, attn_p write, fused P@V ----------------
// One wave handles (bh, 16-row tile). Pass1: online softmax stats via WMMA QK^T.
// Pass2: recompute S, write fp32 P to attn_p, stage bf16 P tile in LDS, WMMA P@V.
__global__ void mha_attn(const __bf16* __restrict__ Qh, const __bf16* __restrict__ Kh,
                         const __bf16* __restrict__ Vt,
                         float* __restrict__ attnp, __bf16* __restrict__ attnb) {
  __shared__ alignas(16) __bf16 ptile[4][16 * 32];   // per-wave 16x32 bf16 P staging

  int lane = threadIdx.x & 31;
  int wave = threadIdx.x >> 5;
  int w = blockIdx.x * 4 + wave;        // 4096 waves total
  int bh = w >> 7;                      // / (T_LEN/16)
  int mTile = w & 127;
  int m0 = mTile * 16;

  const __bf16* Q = Qh + (size_t)bh * T_LEN * HD;
  const __bf16* K = Kh + (size_t)bh * T_LEN * HD;
  const __bf16* V = Vt + (size_t)bh * HD * T_LEN;
  float* P = attnp + (size_t)bh * T_LEN * T_LEN;

  v16bf aq0 = load_a(Q, HD, m0, 0, lane);
  v16bf aq1 = load_a(Q, HD, m0, 32, lane);

  int half = lane >> 4, nl = lane & 15;

  float rm[8], rs[8];
#pragma unroll
  for (int r = 0; r < 8; ++r) { rm[r] = -__builtin_inff(); rs[r] = 0.0f; }

  // ---- pass 1: online softmax statistics over unmasked tiles ----
  for (int nt = 0; nt <= mTile; ++nt) {
    int n0 = nt * 16;
    v16bf bk0 = load_b(K, HD, n0, 0, lane);
    v16bf bk1 = load_b(K, HD, n0, 32, lane);
    v8f s = {};
    s = wmma_bf16(aq0, bk0, s);
    s = wmma_bf16(aq1, bk1, s);
#pragma unroll
    for (int r = 0; r < 8; ++r) {
      int row = m0 + r + 8 * half;
      int col = n0 + nl;
      float sv = (col <= row) ? s[r] : NEG_BIG;
      if (sv > rm[r]) { rs[r] *= __expf(rm[r] - sv); rm[r] = sv; }
      rs[r] += __expf(sv - rm[r]);
    }
  }
  // butterfly combine across the 16 lanes holding each row
#pragma unroll
  for (int r = 0; r < 8; ++r) {
#pragma unroll
    for (int off = 1; off < 16; off <<= 1) {
      float om = __shfl_xor(rm[r], off, 16);
      float os = __shfl_xor(rs[r], off, 16);
      float nm = fmaxf(rm[r], om);
      rs[r] = rs[r] * __expf(rm[r] - nm) + os * __expf(om - nm);
      rm[r] = nm;
    }
  }
  float inv[8];
#pragma unroll
  for (int r = 0; r < 8; ++r) inv[r] = 1.0f / rs[r];

  // ---- pass 2: P writeout + fused P@V ----
  v8f o0 = {}, o1 = {}, o2 = {}, o3 = {};
  __bf16* myp = ptile[wave];
  int chunkEnd = (m0 + 15) >> 5;        // last 32-col chunk with any unmasked entry
  for (int cc = 0; cc <= chunkEnd; ++cc) {
    int c0 = cc * 32;
#pragma unroll
    for (int sub = 0; sub < 2; ++sub) {
      int n0 = c0 + sub * 16;
      if (n0 > m0 + 15) {               // fully masked 16x16 subtile: exact zeros
#pragma unroll
        for (int r = 0; r < 8; ++r) {
          int row = m0 + r + 8 * half;
          P[(size_t)row * T_LEN + n0 + nl] = 0.0f;
          myp[(r + 8 * half) * 32 + sub * 16 + nl] = f2bf(0.0f);
        }
      } else {
        v16bf bk0 = load_b(K, HD, n0, 0, lane);
        v16bf bk1 = load_b(K, HD, n0, 32, lane);
        v8f s = {};
        s = wmma_bf16(aq0, bk0, s);
        s = wmma_bf16(aq1, bk1, s);
#pragma unroll
        for (int r = 0; r < 8; ++r) {
          int row = m0 + r + 8 * half;
          int col = n0 + nl;
          float sv = (col <= row) ? s[r] : NEG_BIG;
          float pv = __expf(sv - rm[r]) * inv[r];
          P[(size_t)row * T_LEN + col] = pv;
          myp[(r + 8 * half) * 32 + sub * 16 + nl] = f2bf(pv);
        }
      }
    }
    // reload P tile (16x32) from LDS in A-operand layout (ds ops in-order per wave)
    v16bf ap = load_a(myp, 32, 0, 0, lane);
    v16bf bv0 = load_b(V, T_LEN, 0,  c0, lane);
    v16bf bv1 = load_b(V, T_LEN, 16, c0, lane);
    v16bf bv2 = load_b(V, T_LEN, 32, c0, lane);
    v16bf bv3 = load_b(V, T_LEN, 48, c0, lane);
    o0 = wmma_bf16(ap, bv0, o0);
    o1 = wmma_bf16(ap, bv1, o1);
    o2 = wmma_bf16(ap, bv2, o2);
    o3 = wmma_bf16(ap, bv3, o3);
  }

  // zero-fill rest of attn_p rows (upper triangle beyond computed chunks)
  int zstart = (chunkEnd + 1) * 32;
  if (zstart < T_LEN) {
    int width = T_LEN - zstart;         // multiple of 32
    int total = (16 * width) >> 2;      // float4 stores
    float4 z4 = make_float4(0.f, 0.f, 0.f, 0.f);
    for (int idx = lane; idx < total; idx += 32) {
      int q = idx * 4;
      int row = q / width;
      int col = zstart + (q % width);
      *(float4*)(&P[(size_t)(m0 + row) * T_LEN + col]) = z4;
    }
  }

  // epilogue: O tile (16 rows x 64 cols) -> bf16 attn buffer [T*B, EMB]
  int b_ = bh >> 3, hh = bh & 7;
#pragma unroll
  for (int r = 0; r < 8; ++r) {
    int t = m0 + r + 8 * half;
    size_t base = ((size_t)t * BSZ + b_) * EMB + hh * HD + nl;
    attnb[base + 0]  = f2bf(o0[r]);
    attnb[base + 16] = f2bf(o1[r]);
    attnb[base + 32] = f2bf(o2[r]);
    attnb[base + 48] = f2bf(o3[r]);
  }
}

// ---------------- output projection ----------------
// out[m, f] = sum_e attnb[m,e] * Wout[f,e] + bout[f]
__global__ void mha_out_gemm(const __bf16* __restrict__ A, const __bf16* __restrict__ Bw,
                             const float* __restrict__ bias, float* __restrict__ C) {
  int lane = threadIdx.x & 31;
  int wave = threadIdx.x >> 5;
  int m0 = blockIdx.x * 16;
  int n0 = (blockIdx.y * 4 + wave) * 16;

  v8f c = {};
  for (int k0 = 0; k0 < EMB; k0 += 32) {
    v16bf a = load_a(A, EMB, m0, k0, lane);
    v16bf b = load_b(Bw, EMB, n0, k0, lane);
    c = wmma_bf16(a, b, c);
  }
  int half = lane >> 4, nl = lane & 15;
  int f = n0 + nl;
  float bv = bias[f];
#pragma unroll
  for (int r = 0; r < 8; ++r) {
    int mrow = m0 + r + 8 * half;
    C[(size_t)mrow * EMB + f] = c[r] + bv;
  }
}

extern "C" void kernel_launch(void* const* d_in, const int* in_sizes, int n_in,
                              void* d_out, int out_size, void* d_ws, size_t ws_size,
                              hipStream_t stream) {
  (void)in_sizes; (void)n_in; (void)out_size; (void)ws_size;
  const float* h      = (const float*)d_in[0];
  // d_in[1] = attn_mask (causal, regenerated analytically — unused)
  const float* wqkv_f = (const float*)d_in[2];
  const float* bqkv   = (const float*)d_in[3];
  const float* wout_f = (const float*)d_in[4];
  const float* bout   = (const float*)d_in[5];

  float* out   = (float*)d_out;
  float* attnp = out + (size_t)MROWS * EMB;       // attn_p region of d_out

  __bf16* ws = (__bf16*)d_ws;
  __bf16* hb = ws;                                // [8192, 512]
  __bf16* wq = hb + (size_t)MROWS * EMB;          // [1536, 512]
  __bf16* wo = wq + (size_t)3 * EMB * EMB;        // [512, 512]
  __bf16* Qh = wo + (size_t)EMB * EMB;            // [32, 2048, 64]
  __bf16* Kh = Qh + (size_t)NBH * T_LEN * HD;     // [32, 2048, 64]
  __bf16* Vt = Kh + (size_t)NBH * T_LEN * HD;     // [32, 64, 2048]
  __bf16* ab = Vt + (size_t)NBH * T_LEN * HD;     // [8192, 512]

  int n1 = MROWS * EMB;
  int n2 = 3 * EMB * EMB;
  int n3 = EMB * EMB;
  mha_cvt_bf16<<<(n1 / 4 + 255) / 256, 256, 0, stream>>>(h, hb, n1);
  mha_cvt_bf16<<<(n2 / 4 + 255) / 256, 256, 0, stream>>>(wqkv_f, wq, n2);
  mha_cvt_bf16<<<(n3 / 4 + 255) / 256, 256, 0, stream>>>(wout_f, wo, n3);

  mha_qkv_gemm<<<dim3(MROWS / 16, (3 * EMB) / 64), 128, 0, stream>>>(hb, wq, bqkv, Qh, Kh, Vt);
  mha_attn<<<(NBH * (T_LEN / 16)) / 4, 128, 0, stream>>>(Qh, Kh, Vt, attnp, ab);
  mha_out_gemm<<<dim3(MROWS / 16, EMB / 16 / 4), 128, 0, stream>>>(ab, wo, bout, out);
}
